// SpatialAttention_65592740544777
// MI455X (gfx1250) — matrix operand order repaired and checked by reading, not compiled, not run on hardware
//
#include <hip/hip_runtime.h>

typedef __attribute__((ext_vector_type(16))) _Float16 v16h;
typedef __attribute__((ext_vector_type(8)))  _Float16 v8h;
typedef __attribute__((ext_vector_type(8)))  float    v8f;

#define BT_    48
#define NN     500
#define NPAD   512
#define HEADS_ 8
#define HD_    16
#define HID_   128

union V16 { v16h h16; v8h p[2]; unsigned u[8]; };
union H2  { _Float16 h[2]; unsigned u; };

// One workgroup (256 threads = 8 waves) per (bt, h). Flash-style one-pass
// masked softmax, transposed orientation: S'[m][n] = K·Q^T so each lane owns
// one query column; O^T[d][n] accumulated directly. All GEMMs on
// v_wmma_f32_16x16x32_f16 with f32 accumulation.
__global__ __launch_bounds__(256) void spattn_flash_kernel(
    const float* __restrict__ x, const int* __restrict__ adj,
    const float* __restrict__ Wq, const float* __restrict__ bq,
    const float* __restrict__ Wk, const float* __restrict__ bk,
    const float* __restrict__ Wv, const float* __restrict__ bv,
    float* __restrict__ ws)
{
    __shared__ _Float16 qh[NPAD][HD_];   // q rows (f16)
    __shared__ _Float16 kh[NPAD][HD_];   // k rows (f16)
    __shared__ _Float16 vT[HD_][NPAD];   // v transposed (f16)
    __shared__ float    sW[3][HD_][HD_];
    __shared__ float    sb[3][HD_];
    __shared__ _Float16 zrow[32];        // zero row for hi-lane B reads

    const int bt  = blockIdx.x >> 3;
    const int h   = blockIdx.x & 7;
    const int tid = threadIdx.x;

    // ---- weights + zero row to LDS ----
    for (int i = tid; i < 3 * 256; i += 256) {
        const float* src = (i < 256) ? Wq : ((i < 512) ? Wk : Wv);
        ((float*)sW)[i] = src[i & 255];
    }
    if (tid < 48) {
        const float* src = (tid < 16) ? bq : ((tid < 32) ? bk : bv);
        ((float*)sb)[tid] = src[tid & 15];
    }
    if (tid < 32) zrow[tid] = (_Float16)0.f;
    __syncthreads();

    // ---- q/k/v projections: f32 FMA -> f16 LDS (vectorized row stores) ----
    for (int n = tid; n < NPAD; n += 256) {
        float xv[16];
        if (n < NN) {
            const float* xp = x + ((size_t)bt * NN + n) * HID_ + h * HD_;
            #pragma unroll
            for (int d = 0; d < 16; ++d) xv[d] = xp[d];
        } else {
            #pragma unroll
            for (int d = 0; d < 16; ++d) xv[d] = 0.f;
        }
        v8h q0, q1, k0, k1;
        #pragma unroll
        for (int d = 0; d < 16; ++d) {
            float aq = sb[0][d], ak = sb[1][d], av = sb[2][d];
            #pragma unroll
            for (int e = 0; e < 16; ++e) {
                aq = fmaf(xv[e], sW[0][d][e], aq);
                ak = fmaf(xv[e], sW[1][d][e], ak);
                av = fmaf(xv[e], sW[2][d][e], av);
            }
            if (n >= NN) { aq = 0.f; ak = 0.f; av = 0.f; }
            if (d < 8) { q0[d] = (_Float16)aq; k0[d] = (_Float16)ak; }
            else       { q1[d - 8] = (_Float16)aq; k1[d - 8] = (_Float16)ak; }
            vT[d][n] = (_Float16)av;     // transpose scatter (u16)
        }
        *(v8h*)&qh[n][0] = q0;  *(v8h*)&qh[n][8] = q1;
        *(v8h*)&kh[n][0] = k0;  *(v8h*)&kh[n][8] = k1;
    }
    __syncthreads();

    const int  w    = tid >> 5;
    const int  lane = tid & 31;
    const int  lh   = lane & 15;
    const bool hi   = lane >= 16;
    const int  roff = hi ? 8 : 0;

    v8h zh;
    #pragma unroll
    for (int j = 0; j < 8; ++j) zh[j] = (_Float16)0.f;

    for (int nt = w; nt < 32; nt += 8) {             // 32 query tiles of 16
        const int  n0   = nt * 16;
        const int  n_un = n0 + lh;                   // this lane's query
        const bool nok  = n_un < NN;
        const int  n_cl = nok ? n_un : (NN - 1);     // clamped for adj reads

        // B_q fragment (loop-invariant): halves j<8 -> K=roff+j of q row; j>=8 zeros
        const _Float16* qsrc = &qh[n_un][roff];
        V16 Bq;
        Bq.p[0] = *(const v8h*)qsrc;
        Bq.p[1] = zh;

        v8f acc;
        #pragma unroll
        for (int r = 0; r < 8; ++r) acc[r] = 0.f;
        float mrun = -__builtin_inff();
        float lsum = 0.f;

        for (int mt = 0; mt < 32; ++mt) {            // 32 key tiles of 16
            const int m0 = mt * 16;

            // A_k fragment: lane row m0+lh, halves j<8 -> K=roff+j
            V16 Ak;
            Ak.p[0] = *(const v8h*)&kh[m0 + lh][roff];
            Ak.p[1] = zh;

            v8f z8;
            #pragma unroll
            for (int r = 0; r < 8; ++r) z8[r] = 0.f;
            v8f s = __builtin_amdgcn_wmma_f32_16x16x32_f16(
                        false, Ak.h16, false, Bq.h16, (short)0, z8, false, false);

            // adjacency row segment: 8 consecutive ints (two int4)
            int arr[8];
            const int* ap = adj + (size_t)n_cl * NN + m0 + roff;
            *(int4*)&arr[0] = *(const int4*)ap;
            *(int4*)&arr[4] = *(const int4*)(ap + 4);

            float val[8];
            #pragma unroll
            for (int r = 0; r < 8; ++r) {
                const int m = m0 + roff + r;
                float v = s[r] * 0.25f;                  // 1/sqrt(16)
                if (m >= NN) v = -__builtin_inff();      // pad keys: weight 0
                else if (!nok || arr[r] == 0) v = -1e9f; // adjacency mask
                val[r] = v;
            }

            // online softmax (per-lane scalars; one cross-half shuffle each)
            float tmax = val[0];
            #pragma unroll
            for (int r = 1; r < 8; ++r) tmax = fmaxf(tmax, val[r]);
            tmax = fmaxf(tmax, __shfl_xor(tmax, 16, 32));
            const float mnew = fmaxf(mrun, tmax);
            const float sc   = __expf(mrun - mnew);
            float p[8], psum = 0.f;
            #pragma unroll
            for (int r = 0; r < 8; ++r) { p[r] = __expf(val[r] - mnew); psum += p[r]; }
            psum += __shfl_xor(psum, 16, 32);
            lsum = lsum * sc + psum;
            #pragma unroll
            for (int r = 0; r < 8; ++r) acc[r] *= sc;
            mrun = mnew;

            // build B_p: pack p to f16x2 dwords; own half used as is (shuffled po unused)
            unsigned pk[4], po[4];
            #pragma unroll
            for (int i = 0; i < 4; ++i) {
                H2 t;
                t.h[0] = (_Float16)p[2 * i];
                t.h[1] = (_Float16)p[2 * i + 1];
                pk[i] = t.u;
            }
            #pragma unroll
            for (int i = 0; i < 4; ++i)
                po[i] = (unsigned)__shfl_xor((int)pk[i], 16, 32);
            V16 Bp;
            #pragma unroll
            for (int i = 0; i < 4; ++i) {
                Bp.u[i]     = pk[i];             // K=roff+0..7 : own m rows
                Bp.u[4 + i] = 0u;                // unused K slots : zero
            }

            // A_v fragment: lane row d=lh, halves j<8 -> K=roff+j
            V16 Av;
            Av.p[0] = *(const v8h*)&vT[lh][m0 + roff];
            Av.p[1] = zh;

            acc = __builtin_amdgcn_wmma_f32_16x16x32_f16(
                      false, Av.h16, false, Bp.h16, (short)0, acc, false, false);
        }

        // epilogue: O[n][d] = acc[d]/lsum ; lane owns 8 contiguous d values
        if (nok) {
            const float rec = 1.f / lsum;
            float* op = ws + (((size_t)bt * HEADS_ + h) * NN + n_un) * HD_ + roff;
            float4 o0, o1;
            o0.x = acc[0] * rec; o0.y = acc[1] * rec;
            o0.z = acc[2] * rec; o0.w = acc[3] * rec;
            o1.x = acc[4] * rec; o1.y = acc[5] * rec;
            o1.z = acc[6] * rec; o1.w = acc[7] * rec;
            *(float4*)op = o0;
            *(float4*)(op + 4) = o1;
        }
    }
}

// Output projection: merged (view of ws as [bt][n*128+c]) @ Wo^T + bo.
// One workgroup per bt; 8 waves x 4 row tiles x 8 col tiles, K=128.
__global__ __launch_bounds__(256) void spattn_proj_kernel(
    const float* __restrict__ ws, const float* __restrict__ Wo,
    const float* __restrict__ bo, float* __restrict__ out)
{
    __shared__ _Float16 wof[HID_][HID_];   // Wo row-major f16 (32 KB)

    const int bt  = blockIdx.x;
    const int tid = threadIdx.x;
    for (int i = tid; i < HID_ * HID_; i += 256)
        ((_Float16*)wof)[i] = (_Float16)Wo[i];
    __syncthreads();

    const int  w    = tid >> 5;
    const int  lane = tid & 31;
    const int  lh   = lane & 15;
    const bool hi   = lane >= 16;
    const int  roff = hi ? 8 : 0;

    const float* mrg = ws + (size_t)bt * (NN * HID_);   // exact reshape view

    for (int rt = w; rt < 32; rt += 8) {
        const int row  = rt * 16 + lh;
        const int row2 = (row < NN) ? row : (NN - 1);   // clamp (rows>=NN unsaved)
        const float* rbase = mrg + (size_t)row2 * HID_;

        v16h a4[4];
        #pragma unroll
        for (int kt = 0; kt < 4; ++kt) {
            const float* c0 = rbase + kt * 32 + roff;        // K chunk 0 (8 f32)
            const float* c1 = c0 + 16;                       // K chunk 1 (8 f32)
            float4 f0 = *(const float4*)c0, f1 = *(const float4*)(c0 + 4);
            float4 g0 = *(const float4*)c1, g1 = *(const float4*)(c1 + 4);
            V16 A;
            A.h16[0]  = (_Float16)f0.x; A.h16[1]  = (_Float16)f0.y;
            A.h16[2]  = (_Float16)f0.z; A.h16[3]  = (_Float16)f0.w;
            A.h16[4]  = (_Float16)f1.x; A.h16[5]  = (_Float16)f1.y;
            A.h16[6]  = (_Float16)f1.z; A.h16[7]  = (_Float16)f1.w;
            A.h16[8]  = (_Float16)g0.x; A.h16[9]  = (_Float16)g0.y;
            A.h16[10] = (_Float16)g0.z; A.h16[11] = (_Float16)g0.w;
            A.h16[12] = (_Float16)g1.x; A.h16[13] = (_Float16)g1.y;
            A.h16[14] = (_Float16)g1.z; A.h16[15] = (_Float16)g1.w;
            a4[kt] = A.h16;
        }

        #pragma unroll
        for (int ct = 0; ct < 8; ++ct) {
            const int   cout = ct * 16 + lh;
            const float bias = bo[cout];
            v8f acc;
            #pragma unroll
            for (int r = 0; r < 8; ++r) acc[r] = bias;   // fold bias into C

            #pragma unroll
            for (int kt = 0; kt < 4; ++kt) {
                // B[k][cout] = Wo[cout][k]; two 16B chunks at k=roff and k=16+roff
                const v8h* wp = (const v8h*)&wof[cout][kt * 32 + roff];
                V16 Bf;
                Bf.p[0] = wp[0];
                Bf.p[1] = wp[2];
                acc = __builtin_amdgcn_wmma_f32_16x16x32_f16(
                          false, a4[kt], false, Bf.h16, (short)0, acc, false, false);
            }

            #pragma unroll
            for (int r = 0; r < 8; ++r) {
                const int n = rt * 16 + roff + r;
                if (n < NN)
                    out[((size_t)bt * NN + n) * HID_ + cout] = acc[r];
            }
        }
    }
}

extern "C" void kernel_launch(void* const* d_in, const int* in_sizes, int n_in,
                              void* d_out, int out_size, void* d_ws, size_t ws_size,
                              hipStream_t stream) {
    const float* x   = (const float*)d_in[0];
    const int*   adj = (const int*)d_in[1];
    const float* Wq  = (const float*)d_in[2];
    const float* bq  = (const float*)d_in[3];
    const float* Wk  = (const float*)d_in[4];
    const float* bk  = (const float*)d_in[5];
    const float* Wv  = (const float*)d_in[6];
    const float* bv  = (const float*)d_in[7];
    const float* Wo  = (const float*)d_in[8];
    const float* bo  = (const float*)d_in[9];
    float* out = (float*)d_out;
    float* ws  = (float*)d_ws;   // needs BT*H*N*HD floats = 12.3 MB

    spattn_flash_kernel<<<dim3(BT_ * HEADS_), dim3(256), 0, stream>>>(
        x, adj, Wq, bq, Wk, bk, Wv, bv, ws);
    spattn_proj_kernel<<<dim3(BT_), dim3(256), 0, stream>>>(ws, Wo, bo, out);
}
